// MSDMSA_bottle2neck_6038724018804
// MI455X (gfx1250) — matrix operand order repaired and checked by reading, not compiled, not run on hardware
//
#include <hip/hip_runtime.h>
#include <cstdint>

// ---------------------------------------------------------------------------
// Types for gfx1250 WMMA (wave32). bf16 A/B fragments = 16 x __bf16 (8 VGPRs),
// f32 C/D accumulator = 8 floats (8 VGPRs).
// ---------------------------------------------------------------------------
typedef __attribute__((ext_vector_type(16))) __bf16 v16bf;
typedef __attribute__((ext_vector_type(8)))  __bf16 v8bf;
typedef __attribute__((ext_vector_type(8)))  float  v8f;
typedef __attribute__((ext_vector_type(4)))  float  v4f;
typedef __attribute__((ext_vector_type(4)))  unsigned v4u;
typedef __attribute__((ext_vector_type(8)))  unsigned v8u;

static __device__ inline v16bf make_frag(v4u a, v4u b) {
    v8u t = __builtin_shufflevector(a, b, 0, 1, 2, 3, 4, 5, 6, 7);
    return __builtin_bit_cast(v16bf, t);
}

// ---------------------------------------------------------------------------
// Tiled bf16 WMMA GEMM:  C[M,N] = A[M,K] * B[K,N]  (A,B fp32 row-major in
// global, converted to bf16 while staging through double-buffered LDS).
// Block = 256 threads = 8 waves; tile 128x64, K-step 32.
// Wave w: 32x32 sub-tile at rows (w&3)*32, cols (w>>2)*32 -> 2 A-frags x
// 2 B-frags -> 4 WMMAs per K-step (outer-product fragment reuse).
// Staging is branch-free: vectorized b128 loads on interior tiles, clamped
// address + select on edge tiles (no divergent exec-mask serialization).
// EPI: 1 = +bias[n]; 2 = +bias[n] then exact GELU;
//      3 = scale[m]*x+shift[m], ReLU (fused BN);
//      4 = EPI3 + residual add (same layout as C) before ReLU.
// ---------------------------------------------------------------------------
#define BM 128
#define BN 64
#define BKK 32
#define LDSW 40   // bf16 units per LDS row (80 bytes) -> bank-rotating b128 reads

template <int EPI>
__global__ __launch_bounds__(256) void gemm_bf16_wmma(
    const float* __restrict__ A, const float* __restrict__ B,
    float* __restrict__ C, int M, int N, int K, int lda, int ldb, int ldc,
    long long strA, long long strB, long long strC,
    const float* __restrict__ p0, const float* __restrict__ p1,
    const float* __restrict__ res, long long strRes)
{
    __shared__ __align__(16) __bf16 shA[2][BM * LDSW];
    __shared__ __align__(16) __bf16 shB[2][BN * LDSW];

    const int z = blockIdx.z;
    A += (long long)z * strA;
    B += (long long)z * strB;
    C += (long long)z * strC;
    const float* R = (EPI == 4) ? res + (long long)z * strRes : nullptr;

    const int bm = blockIdx.y * BM;
    const int bn = blockIdx.x * BN;
    const int tid = threadIdx.x;
    const int wave = tid >> 5, lane = tid & 31;
    const int mgw = (wave & 3) * 32;   // wave m-origin within tile
    const int ngw = (wave >> 2) * 32;  // wave n-origin within tile
    const int lm = lane & 15, lh = lane >> 4;

    // staging maps
    const int ar = tid >> 1;           // A row 0..127
    const int ac = (tid & 1) * 16;     // A col start (16 consecutive K)
    const int nn = (tid & 15) * 4;     // B: 4 consecutive N
    const int kp = (tid >> 4) * 2;     // B: k-pair 0,2,..,30

    const bool fullM = (bm + BM <= M);
    const bool fullN = (bn + BN <= N);
    const int kTiles = (K + BKK - 1) / BKK;

    v8f acc[2][2] = {};
    float ra[16], rb[8];

    auto loadA = [&](int kt) {
        const int k0 = kt * BKK;
        const int mg = bm + ar;
        if (fullM && (k0 + BKK <= K)) {
            const v4f* p = reinterpret_cast<const v4f*>(A + (long long)mg * lda + k0 + ac);
#pragma unroll
            for (int q = 0; q < 4; ++q) {
                const v4f x = p[q];
#pragma unroll
                for (int i = 0; i < 4; ++i) ra[q * 4 + i] = x[i];
            }
        } else {
            const int mc = (mg < M) ? mg : (M - 1);
#pragma unroll
            for (int i = 0; i < 16; ++i) {
                const int kg = k0 + ac + i;
                const int kc = (kg < K) ? kg : (K - 1);
                const float v = A[(long long)mc * lda + kc];      // always in-bounds
                ra[i] = (mg < M && kg < K) ? v : 0.f;             // branch-free select
            }
        }
    };

    auto loadB = [&](int kt) {
        const int k0 = kt * BKK;
        const int kg0 = k0 + kp, kg1 = kg0 + 1;
        if (fullN && (k0 + BKK <= K)) {
            const v4f x = *reinterpret_cast<const v4f*>(B + (long long)kg0 * ldb + bn + nn);
            const v4f y = *reinterpret_cast<const v4f*>(B + (long long)kg1 * ldb + bn + nn);
#pragma unroll
            for (int j = 0; j < 4; ++j) { rb[j] = x[j]; rb[4 + j] = y[j]; }
        } else {
            const int kc0 = (kg0 < K) ? kg0 : (K - 1);
            const int kc1 = (kg1 < K) ? kg1 : (K - 1);
#pragma unroll
            for (int j = 0; j < 4; ++j) {
                const int ng = bn + nn + j;
                const int nc = (ng < N) ? ng : (N - 1);
                const float v0 = B[(long long)kc0 * ldb + nc];
                const float v1 = B[(long long)kc1 * ldb + nc];
                rb[j]     = (kg0 < K && ng < N) ? v0 : 0.f;
                rb[4 + j] = (kg1 < K && ng < N) ? v1 : 0.f;
            }
        }
    };

    auto storeA = [&](int buf) {
#pragma unroll
        for (int q = 0; q < 2; ++q) {
            v8bf h;
#pragma unroll
            for (int i = 0; i < 8; ++i) h[i] = (__bf16)ra[q * 8 + i];
            *reinterpret_cast<v4u*>(shA[buf] + ar * LDSW + ac + q * 8) =
                __builtin_bit_cast(v4u, h);
        }
    };

    auto storeB = [&](int buf) {   // transposed [n][k], k-pair packed per dword
#pragma unroll
        for (int j = 0; j < 4; ++j) {
            const unsigned short lo = __builtin_bit_cast(unsigned short, (__bf16)rb[j]);
            const unsigned short hi = __builtin_bit_cast(unsigned short, (__bf16)rb[4 + j]);
            const unsigned u = (unsigned)lo | ((unsigned)hi << 16);
            *reinterpret_cast<unsigned*>(shB[buf] + (nn + j) * LDSW + kp) = u;
        }
    };

    // prologue: stage tile 0
    loadA(0); loadB(0);
    storeA(0); storeB(0);
    __syncthreads();

    for (int kt = 0; kt < kTiles; ++kt) {
        const int cur = kt & 1, nxt = cur ^ 1;
        const bool hasNext = (kt + 1 < kTiles);
        if (hasNext) { loadA(kt + 1); loadB(kt + 1); }   // issue global loads early
        if (kt + 2 < kTiles) {                           // speculative L2 prefetch
            __builtin_prefetch(A + (long long)(bm + ar) * lda + (kt + 2) * BKK + ac, 0, 0);
            __builtin_prefetch(B + (long long)((kt + 2) * BKK + kp) * ldb + bn + nn, 0, 0);
        }

        // fragments from current buffer; 4 WMMAs per wave per K-step
        const v4u* pa0 = reinterpret_cast<const v4u*>(shA[cur] + (mgw + lm) * LDSW);
        const v4u* pa1 = reinterpret_cast<const v4u*>(shA[cur] + (mgw + 16 + lm) * LDSW);
        const v4u* pb0 = reinterpret_cast<const v4u*>(shB[cur] + (ngw + lm) * LDSW);
        const v4u* pb1 = reinterpret_cast<const v4u*>(shB[cur] + (ngw + 16 + lm) * LDSW);
        const v16bf fa0 = make_frag(pa0[lh], pa0[2 + lh]);
        const v16bf fa1 = make_frag(pa1[lh], pa1[2 + lh]);
        const v16bf fb0 = make_frag(pb0[lh], pb0[2 + lh]);
        const v16bf fb1 = make_frag(pb1[lh], pb1[2 + lh]);

        acc[0][0] = __builtin_amdgcn_wmma_f32_16x16x32_bf16(false, fa0, false, fb0,
                                                            (short)0, acc[0][0], false, false);
        acc[0][1] = __builtin_amdgcn_wmma_f32_16x16x32_bf16(false, fa0, false, fb1,
                                                            (short)0, acc[0][1], false, false);
        acc[1][0] = __builtin_amdgcn_wmma_f32_16x16x32_bf16(false, fa1, false, fb0,
                                                            (short)0, acc[1][0], false, false);
        acc[1][1] = __builtin_amdgcn_wmma_f32_16x16x32_bf16(false, fa1, false, fb1,
                                                            (short)0, acc[1][1], false, false);

        if (hasNext) { storeA(nxt); storeB(nxt); }       // fill other buffer
        __syncthreads();
    }

    // epilogue + store. C 16x16 f32 layout: n = lane%16, m = (lane/16)*8 + vgpr
#pragma unroll
    for (int tm = 0; tm < 2; ++tm) {
#pragma unroll
        for (int tn = 0; tn < 2; ++tn) {
            const v8f a = acc[tm][tn];
            const int ng = bn + ngw + tn * 16 + lm;
            if (ng < N) {
#pragma unroll
                for (int v = 0; v < 8; ++v) {
                    const int mg = bm + mgw + tm * 16 + lh * 8 + v;
                    if (mg < M) {
                        float r = a[v];
                        if (EPI == 1 || EPI == 2) r += p0[ng];
                        if (EPI == 2) r = 0.5f * r * (1.f + erff(r * 0.70710678118f));
                        if (EPI == 3 || EPI == 4) r = r * p0[mg] + p1[mg];
                        if (EPI == 4) r += R[(long long)mg * ldc + ng];
                        if (EPI == 3 || EPI == 4) r = fmaxf(r, 0.f);
                        C[(long long)mg * ldc + ng] = r;
                    }
                }
            }
        }
    }
}

// ---------------------------------------------------------------------------
// BN folding: s = g/sqrt(v+eps), t = b - m*s
// ---------------------------------------------------------------------------
__global__ void bnprep(const float* g, const float* b, const float* m,
                       const float* v, float* s, float* t, int n)
{
    int i = blockIdx.x * blockDim.x + threadIdx.x;
    if (i < n) {
        float sc = g[i] * rsqrtf(v[i] + 1e-5f);
        s[i] = sc;
        t[i] = b[i] - m[i] * sc;
    }
}

// ---------------------------------------------------------------------------
// im2col for 3x3/pad1 over 56x56, optional fused add of a second input
// (res2net "sp + spx[i]"). out[n][cin*9+ky*3+kx][hw].
// ---------------------------------------------------------------------------
__global__ void im2col3x3(const float* __restrict__ base, const float* __restrict__ addp,
                          float* __restrict__ out, long long strIn, long long strOut)
{
    const long long total = 936LL * 3136LL;
    long long idx = (long long)blockIdx.x * blockDim.x + threadIdx.x;
    if (idx >= total) return;
    const int n = blockIdx.z;
    const int hw = (int)(idx % 3136);
    const int kr = (int)(idx / 3136);
    const int kx = kr % 3, ky = (kr / 3) % 3, cin = kr / 9;
    const int y = hw / 56, x = hw % 56;
    const int iy = y + ky - 1, ix = x + kx - 1;
    float v = 0.f;
    if (iy >= 0 && iy < 56 && ix >= 0 && ix < 56) {
        const long long o = (long long)cin * 3136 + iy * 56 + ix;
        v = base[(long long)n * strIn + o];
        if (addp) v += addp[(long long)n * strIn + o];
    }
    out[(long long)n * strOut + idx] = v;
}

// ---------------------------------------------------------------------------
// Sine positional embedding (normalize, scale=2pi) + level embed.
// lvl_pos[t=l*3136+hw][c], c<52 -> y-part, c>=52 -> x-part.
// ---------------------------------------------------------------------------
__global__ void pos_embed(const float* __restrict__ lvl_emb, float* __restrict__ out)
{
    const long long total = 12544LL * 104LL;
    long long i = (long long)blockIdx.x * blockDim.x + threadIdx.x;
    if (i >= total) return;
    const int c = (int)(i % 104);
    const int t = (int)(i / 104);
    const int l = t / 3136, hw = t % 3136;
    const int y = hw / 56, x = hw % 56;
    int cc = c, pos = y;
    if (c >= 52) { cc = c - 52; pos = x; }
    const float e = (float)(pos + 1) / (56.f + 1e-6f) * 6.2831853071795864769f;
    const float dimt = powf(10000.f, (2.f * (float)(cc >> 1)) / 52.f);
    const float p = e / dimt;
    const float val = (cc & 1) ? cosf(p) : sinf(p);
    out[i] = val + lvl_emb[l * 104 + c];
}

// feats (channel-major, levels 0..2 in featC; level 3 = out1 rows 312..415)
// -> token-major src[n][l*3136+hw][104]
__global__ void pack_src(const float* __restrict__ featC, const float* __restrict__ out1,
                         float* __restrict__ src)
{
    const long long total = 2LL * 12544LL * 104LL;
    long long i = (long long)blockIdx.x * blockDim.x + threadIdx.x;
    if (i >= total) return;
    const int c = (int)(i % 104);
    const int t = (int)((i / 104) % 12544);
    const int n = (int)(i / (104LL * 12544LL));
    const int l = t / 3136, hw = t % 3136;
    float v;
    if (l < 3)
        v = featC[(long long)n * 416 * 3136 + (long long)(l * 104 + c) * 3136 + hw];
    else
        v = out1[(long long)n * 416 * 3136 + (long long)(312 + c) * 3136 + hw];
    src[i] = v;
}

__global__ void add_pos(const float* __restrict__ src, const float* __restrict__ lp,
                        float* __restrict__ q)
{
    const long long total = 2LL * 12544LL * 104LL;
    long long i = (long long)blockIdx.x * blockDim.x + threadIdx.x;
    if (i >= total) return;
    q[i] = src[i] + lp[i % (12544LL * 104LL)];
}

// ---------------------------------------------------------------------------
// Multi-scale deformable sampling. One wave per (n, token, head); lane = head
// channel (26 active of 32). Softmax over the 16 (level,point) logits is
// computed redundantly per lane (tiny), then 16 bilinear gathers from the
// value tensor [n][l*3136+hw][104].
// ---------------------------------------------------------------------------
__global__ __launch_bounds__(256) void msda_sample(
    const float* __restrict__ off, const float* __restrict__ logits,
    const float* __restrict__ val, float* __restrict__ samp)
{
    const int unit = blockIdx.x * 8 + (threadIdx.x >> 5);   // (n*12544+t)*4 + h
    const int lane = threadIdx.x & 31;
    const int h = unit & 3;
    const int nt = unit >> 2;
    const int t = nt % 12544;
    const int n = nt / 12544;
    const int hw = t % 3136;
    const float rx = ((float)(hw % 56) + 0.5f) / 56.f;
    const float ry = ((float)(hw / 56) + 0.5f) / 56.f;

    const float* lg = logits + (long long)nt * 64 + h * 16;
    const float* of = off + (long long)nt * 128 + h * 32;

    float w[16];
    float mx = -1e30f;
#pragma unroll
    for (int i = 0; i < 16; ++i) { w[i] = lg[i]; mx = fmaxf(mx, w[i]); }
    float s = 0.f;
#pragma unroll
    for (int i = 0; i < 16; ++i) { w[i] = __expf(w[i] - mx); s += w[i]; }
    const float inv = 1.f / s;

    const int hc = lane;
    const bool act = hc < 26;
    const float* vbase = val + (long long)n * 12544 * 104 + h * 26 + (act ? hc : 0);
    float acc = 0.f;

#pragma unroll
    for (int l = 0; l < 4; ++l) {
        const float* vl = vbase + (long long)l * 3136 * 104;
#pragma unroll
        for (int p = 0; p < 4; ++p) {
            const float ox = of[(l * 4 + p) * 2 + 0];
            const float oy = of[(l * 4 + p) * 2 + 1];
            const float xp = (rx + ox / 56.f) * 56.f - 0.5f;
            const float yp = (ry + oy / 56.f) * 56.f - 0.5f;
            const float fx0 = floorf(xp), fy0 = floorf(yp);
            const float fx = xp - fx0, fy = yp - fy0;
            const int x0 = (int)fx0, y0 = (int)fy0;
            const float wgt = w[l * 4 + p] * inv;

            float cv[4];
            const int xs[2] = {x0, x0 + 1};
            const int ys[2] = {y0, y0 + 1};
#pragma unroll
            for (int cy = 0; cy < 2; ++cy)
#pragma unroll
                for (int cx = 0; cx < 2; ++cx) {
                    const int xi = xs[cx], yi = ys[cy];
                    const bool ok = (xi >= 0) & (xi < 56) & (yi >= 0) & (yi < 56);
                    const int xc = min(max(xi, 0), 55);
                    const int yc = min(max(yi, 0), 55);
                    const float vv = vl[(long long)(yc * 56 + xc) * 104];
                    cv[cy * 2 + cx] = ok ? vv : 0.f;
                }
            acc += wgt * ((1.f - fx) * (1.f - fy) * cv[0] + fx * (1.f - fy) * cv[1] +
                          (1.f - fx) * fy * cv[2] + fx * fy * cv[3]);
        }
    }
    if (act) samp[(long long)nt * 104 + h * 26 + hc] = acc;
}

// ---------------------------------------------------------------------------
// src = LayerNorm(src + delta) * g + b ; one wave per token (D=104).
// ---------------------------------------------------------------------------
__global__ __launch_bounds__(256) void ln_residual(
    float* __restrict__ src, const float* __restrict__ delta,
    const float* __restrict__ g, const float* __restrict__ b)
{
    const long long t = (long long)blockIdx.x * 8 + (threadIdx.x >> 5);  // 0..25087
    const int lane = threadIdx.x & 31;
    float vals[4];
    float s = 0.f;
    const float* ps = src + t * 104;
    const float* pd = delta + t * 104;
#pragma unroll
    for (int i = 0; i < 4; ++i) {
        const int c = lane + i * 32;
        const float v = (c < 104) ? (ps[c] + pd[c]) : 0.f;
        vals[i] = v;
        s += v;
    }
#pragma unroll
    for (int o = 16; o; o >>= 1) s += __shfl_xor(s, o, 32);
    const float mean = s * (1.f / 104.f);
    float vs = 0.f;
#pragma unroll
    for (int i = 0; i < 4; ++i) {
        const int c = lane + i * 32;
        const float d = (c < 104) ? (vals[i] - mean) : 0.f;
        vs += d * d;
    }
#pragma unroll
    for (int o = 16; o; o >>= 1) vs += __shfl_xor(vs, o, 32);
    const float r = rsqrtf(vs * (1.f / 104.f) + 1e-5f);
    float* po = src + t * 104;
#pragma unroll
    for (int i = 0; i < 4; ++i) {
        const int c = lane + i * 32;
        if (c < 104) po[c] = (vals[i] - mean) * r * g[c] + b[c];
    }
}

// src tokens -> conv3 input: out[n][c*4+l][hw]
__global__ void pack_conv3(const float* __restrict__ src, float* __restrict__ out)
{
    const long long total = 2LL * 416LL * 3136LL;
    long long i = (long long)blockIdx.x * blockDim.x + threadIdx.x;
    if (i >= total) return;
    const int hw = (int)(i % 3136);
    const int ch = (int)((i / 3136) % 416);
    const int n = (int)(i / (3136LL * 416LL));
    const int l = ch & 3, c = ch >> 2;
    out[i] = src[((long long)n * 12544 + (long long)l * 3136 + hw) * 104 + c];
}

// ---------------------------------------------------------------------------
// host-side GEMM dispatch
// ---------------------------------------------------------------------------
static void launch_gemm(int epi, const float* A, const float* B, float* C,
                        int M, int N, int K, int lda, int ldb, int ldc,
                        long long sA, long long sB, long long sC,
                        const float* p0, const float* p1,
                        const float* res, long long sRes, int batches,
                        hipStream_t st)
{
    dim3 grid((N + BN - 1) / BN, (M + BM - 1) / BM, batches), blk(256);
    switch (epi) {
        case 1: gemm_bf16_wmma<1><<<grid, blk, 0, st>>>(A, B, C, M, N, K, lda, ldb, ldc, sA, sB, sC, p0, p1, res, sRes); break;
        case 2: gemm_bf16_wmma<2><<<grid, blk, 0, st>>>(A, B, C, M, N, K, lda, ldb, ldc, sA, sB, sC, p0, p1, res, sRes); break;
        case 3: gemm_bf16_wmma<3><<<grid, blk, 0, st>>>(A, B, C, M, N, K, lda, ldb, ldc, sA, sB, sC, p0, p1, res, sRes); break;
        case 4: gemm_bf16_wmma<4><<<grid, blk, 0, st>>>(A, B, C, M, N, K, lda, ldb, ldc, sA, sB, sC, p0, p1, res, sRes); break;
    }
}

extern "C" void kernel_launch(void* const* d_in, const int* in_sizes, int n_in,
                              void* d_out, int out_size, void* d_ws, size_t ws_size,
                              hipStream_t stream)
{
    (void)in_sizes; (void)n_in; (void)out_size; (void)ws_size;

    const float* X      = (const float*)d_in[0];   // [2,1024,56,56]
    const float* conv1w = (const float*)d_in[1];   // [416,1024]
    const float* bn1g = (const float*)d_in[2], *bn1b = (const float*)d_in[3];
    const float* bn1m = (const float*)d_in[4], *bn1v = (const float*)d_in[5];
    const float* convsw = (const float*)d_in[6];   // [3,104,936]
    const float* bnsg = (const float*)d_in[7], *bnsb = (const float*)d_in[8];
    const float* bnsm = (const float*)d_in[9], *bnsv = (const float*)d_in[10];
    const float* conv3w = (const float*)d_in[11];  // [1024,416]
    const float* bn3g = (const float*)d_in[12], *bn3b = (const float*)d_in[13];
    const float* bn3m = (const float*)d_in[14], *bn3v = (const float*)d_in[15];
    const float* lvlemb = (const float*)d_in[16];  // [4,104]
    auto LP = [&](int li, int j) { return (const float*)d_in[17 + li * 16 + j]; };

    float* W = (float*)d_ws;
    // workspace arena (float offsets)
    const long long f_out1   = 0;                         // 2*416*3136
    const long long f_featC  = f_out1   + 2609152LL;      // 2*416*3136
    const long long f_im2col = f_featC  + 2609152LL;      // 2*936*3136 (also conv3 input later)
    const long long f_src    = f_im2col + 5870592LL;      // 2*12544*104
    const long long f_lvlpos = f_src    + 2609152LL;      // 12544*104
    const long long f_q      = f_lvlpos + 1304576LL;      // 2*12544*104
    const long long f_val    = f_q      + 2609152LL;      // 2*12544*104
    const long long f_off    = f_val    + 2609152LL;      // 2*12544*128
    const long long f_attn   = f_off    + 3211264LL;      // 2*12544*64
    const long long f_samp   = f_attn   + 1605632LL;      // 2*12544*104
    const long long f_ffh    = f_samp   + 2609152LL;      // 2*12544*208
    const long long f_tmp    = f_ffh    + 5218304LL;      // 2*12544*104
    const long long f_bn     = f_tmp    + 2609152LL;      // folded BN params

    float* bnS1 = W + f_bn;
    float* bnT1 = bnS1 + 416;
    float* bnSs = bnT1 + 416;
    float* bnTs = bnSs + 312;
    float* bnS3 = bnTs + 312;
    float* bnT3 = bnS3 + 1024;

    bnprep<<<2, 256, 0, stream>>>(bn1g, bn1b, bn1m, bn1v, bnS1, bnT1, 416);
    bnprep<<<2, 256, 0, stream>>>(bnsg, bnsb, bnsm, bnsv, bnSs, bnTs, 312);
    bnprep<<<4, 256, 0, stream>>>(bn3g, bn3b, bn3m, bn3v, bnS3, bnT3, 1024);

    // conv1 (1x1) + BN + ReLU : out1 = relu(bn(conv1_w @ x))
    launch_gemm(3, conv1w, X, W + f_out1, 416, 3136, 1024, 1024, 3136, 3136,
                0, 1024LL * 3136, 416LL * 3136, bnS1, bnT1, nullptr, 0, 2, stream);

    // res2net chain: 3x (im2col + GEMM + BN + ReLU)
    for (int i = 0; i < 3; ++i) {
        const float* base = (i == 0) ? (W + f_out1)
                                     : (W + f_featC + (long long)(i - 1) * 104 * 3136);
        const float* addp = (i == 0) ? nullptr
                                     : (W + f_out1 + (long long)i * 104 * 3136);
        const long long tot = 936LL * 3136LL;
        im2col3x3<<<dim3((unsigned)((tot + 255) / 256), 1, 2), 256, 0, stream>>>(
            base, addp, W + f_im2col, 416LL * 3136, 936LL * 3136);
        launch_gemm(3, convsw + (long long)i * 104 * 936, W + f_im2col,
                    W + f_featC + (long long)i * 104 * 3136,
                    104, 3136, 936, 936, 3136, 3136,
                    0, 936LL * 3136, 416LL * 3136,
                    bnSs + i * 104, bnTs + i * 104, nullptr, 0, 2, stream);
    }

    pack_src<<<10192, 256, 0, stream>>>(W + f_featC, W + f_out1, W + f_src);
    pos_embed<<<5096, 256, 0, stream>>>(lvlemb, W + f_lvlpos);

    const long long sTok = 12544LL * 104;
    for (int li = 0; li < 3; ++li) {
        add_pos<<<10192, 256, 0, stream>>>(W + f_src, W + f_lvlpos, W + f_q);
        // sampling offsets: q @ ow + ob
        launch_gemm(1, W + f_q, LP(li, 0), W + f_off, 12544, 128, 104, 104, 128, 128,
                    sTok, 0, 12544LL * 128, LP(li, 1), nullptr, nullptr, 0, 2, stream);
        // attention logits: q @ aw + ab  (softmax fused into sampler)
        launch_gemm(1, W + f_q, LP(li, 2), W + f_attn, 12544, 64, 104, 104, 64, 64,
                    sTok, 0, 12544LL * 64, LP(li, 3), nullptr, nullptr, 0, 2, stream);
        // value projection: src @ vw + vb
        launch_gemm(1, W + f_src, LP(li, 4), W + f_val, 12544, 104, 104, 104, 104, 104,
                    sTok, 0, sTok, LP(li, 5), nullptr, nullptr, 0, 2, stream);
        msda_sample<<<12544, 256, 0, stream>>>(W + f_off, W + f_attn, W + f_val, W + f_samp);
        // output projection
        launch_gemm(1, W + f_samp, LP(li, 6), W + f_tmp, 12544, 104, 104, 104, 104, 104,
                    sTok, 0, sTok, LP(li, 7), nullptr, nullptr, 0, 2, stream);
        ln_residual<<<3136, 256, 0, stream>>>(W + f_src, W + f_tmp, LP(li, 12), LP(li, 13));
        // FFN: f1 (+GELU fused), f2
        launch_gemm(2, W + f_src, LP(li, 8), W + f_ffh, 12544, 208, 104, 104, 208, 208,
                    sTok, 0, 12544LL * 208, LP(li, 9), nullptr, nullptr, 0, 2, stream);
        launch_gemm(1, W + f_ffh, LP(li, 10), W + f_tmp, 12544, 104, 208, 208, 104, 104,
                    12544LL * 208, 0, sTok, LP(li, 11), nullptr, nullptr, 0, 2, stream);
        ln_residual<<<3136, 256, 0, stream>>>(W + f_src, W + f_tmp, LP(li, 14), LP(li, 15));
    }

    // conv3 (1x1) + BN + residual(x) + ReLU -> d_out
    pack_conv3<<<10192, 256, 0, stream>>>(W + f_src, W + f_im2col);
    launch_gemm(4, conv3w, W + f_im2col, (float*)d_out, 1024, 3136, 416, 416, 3136, 3136,
                0, 416LL * 3136, 1024LL * 3136, bnS3, bnT3, X, 1024LL * 3136, 2, stream);
}
